// IDetectWithAttributes_55757265437059
// MI455X (gfx1250) — matrix operand order repaired and checked by reading, not compiled, not run on hardware
//
#include <hip/hip_runtime.h>
#include <math.h>

typedef __attribute__((ext_vector_type(16))) __bf16 v16bf;
typedef __attribute__((ext_vector_type(8)))  float  v8f;

#define KNO 105
#define KNA 3
#define KNOA 315
#define OTILES 20
#define OPAD (OTILES * 16)   /* 320 */
#define POSB 64
#define NTHREADS 128
#define FSTRIDE 20           /* fragment row stride in dwords: 16B-aligned + conflict-free */

__device__ __forceinline__ unsigned pack2bf16(float a, float b) {
  unsigned ua = __float_as_uint(a);
  unsigned ub = __float_as_uint(b);
  ua = (ua + 0x7FFFu + ((ua >> 16) & 1u)) >> 16;
  ub = (ub + 0x7FFFu + ((ub >> 16) & 1u)) >> 16;
  return (ua & 0xFFFFu) | (ub << 16);
}

__device__ __forceinline__ float sigmoidf_(float x) {
  return 1.0f / (1.0f + __expf(-x));
}

__global__ __launch_bounds__(NTHREADS)
void idetect_level_kernel(const float* __restrict__ x,
                          const float* __restrict__ wmat,
                          const float* __restrict__ bias,
                          const float* __restrict__ ia,
                          const float* __restrict__ im,
                          float* __restrict__ out,
                          int C, int HW, int nxdim, float stride,
                          float aw0, float ah0, float aw1, float ah1,
                          float aw2, float ah2, int qoff, int totalQ)
{
  // Lane-major fragment staging: row = pos/o, 16 k-pair dwords in VGPR order.
  // Row stride 20 dwords: rows stay 16B aligned and 16 lanes' b128 reads are
  // bank-conflict-free (20*m mod 64 distinct for m in [0,16)).
  __shared__ __align__(16) unsigned a_lds[POSB][FSTRIDE];
  __shared__ __align__(16) unsigned b_lds[OPAD][FSTRIDE];
  // f32 GEMM result / decode buffer
  __shared__ float y_lds[POSB][OPAD + 1];
  __shared__ float bb_lds[OPAD];
  __shared__ float mm_lds[OPAD];
  __shared__ float ia_lds[1024];

  const int tid  = threadIdx.x;
  const int lane = tid & 31;
  const int wave = tid >> 5;
  const int half = lane >> 4;     // which 16-lane half of the wave
  const int m    = lane & 15;

  const int pbase = blockIdx.x * POSB;
  const int batch = blockIdx.y;
  const float* xb = x + (size_t)batch * (size_t)C * (size_t)HW;

  for (int i = tid; i < C; i += NTHREADS) ia_lds[i] = ia[i];
  for (int i = tid; i < OPAD; i += NTHREADS) {
    int o = (i < KNOA) ? i : (KNOA - 1);
    bb_lds[i] = bias[o];
    mm_lds[i] = im[o];
  }

  v8f acc[OTILES];
  #pragma unroll
  for (int t = 0; t < OTILES; ++t)
    acc[t] = v8f{0.f, 0.f, 0.f, 0.f, 0.f, 0.f, 0.f, 0.f};

  for (int k0 = 0; k0 < C; k0 += 32) {
    __syncthreads();   // protect LDS staging buffers across iterations
    // ---- stage A: 64 positions x 16 k-pairs, add ImplicitA, cvt bf16.
    // Column = A-fragment VGPR order: swap bits 2<->3 of k-pair index
    // (VGPR 0..3 hold k-pairs {0..3 | +4 for hi half}, VGPR 4..7 hold {8..11 | +4}).
    #pragma unroll
    for (int it = 0; it < (16 * POSB) / NTHREADS; ++it) {
      int flat = it * NTHREADS + tid;
      int pos  = flat & (POSB - 1);
      int j    = flat >> 6;               // k-pair index 0..15
      int col  = (j & 3) | ((j & 8) >> 1) | ((j & 4) << 1);
      int p    = pbase + pos; if (p >= HW) p = HW - 1;   // clamp (no EXEC holes)
      int c0   = k0 + 2 * j;
      const float* xp0 = &xb[(size_t)c0 * HW + p];
      const float* xp1 = &xb[(size_t)(c0 + 1) * HW + p];
      if (k0 + 32 < C) {                  // prefetch next k-chunk of the x stream
        __builtin_prefetch(xp0 + (size_t)32 * HW, 0, 1);
        __builtin_prefetch(xp1 + (size_t)32 * HW, 0, 1);
      }
      float f0 = xp0[0] + ia_lds[c0];
      float f1 = xp1[0] + ia_lds[c0 + 1];
      a_lds[pos][col] = pack2bf16(f0, f1);
    }
    // ---- stage B: 320 outputs x 16 k-pairs (o >= 315 clamped).
    // Column = B-fragment VGPR order: swap bits 2<->3 of k-pair index (same as A).
    for (int it = 0; it < (16 * OPAD) / NTHREADS; ++it) {
      int flat = it * NTHREADS + tid;
      int j    = flat & 15;
      int o    = flat >> 4;
      int oc   = (o < KNOA) ? o : (KNOA - 1);
      const float* wr = wmat + (size_t)oc * C + (k0 + 2 * j);
      b_lds[o][(j & 3) | ((j & 8) >> 1) | ((j & 4) << 1)] = pack2bf16(wr[0], wr[1]);
    }
    __syncthreads();

    // ---- per-wave A fragment: two ds_load_b128 per lane
    union { uint4 q[2]; v16bf v; } afr;
    {
      const uint4* ap = (const uint4*)(&a_lds[wave * 16 + m][half * 8]);
      afr.q[0] = ap[0];
      afr.q[1] = ap[1];
    }
    // ---- 20 WMMAs, one per o-tile; B fragment: two ds_load_b128 per lane
    #pragma unroll
    for (int t = 0; t < OTILES; ++t) {
      union { uint4 q[2]; v16bf v; } bfr;
      const uint4* bp = (const uint4*)(&b_lds[t * 16 + m][half * 8]);
      bfr.q[0] = bp[0];
      bfr.q[1] = bp[1];
      acc[t] = __builtin_amdgcn_wmma_f32_16x16x32_bf16(
          false, afr.v, false, bfr.v, (short)0, acc[t], false, false);
    }
  }

  // ---- epilogue: (acc + bias) * ImplicitM -> y_lds  (C/D layout: M=r+8*half, N=m)
  {
    int posl0 = wave * 16 + half * 8;
    #pragma unroll
    for (int t = 0; t < OTILES; ++t) {
      int o = t * 16 + m;
      float bo = bb_lds[o];
      float mo = mm_lds[o];
      #pragma unroll
      for (int r = 0; r < 8; ++r)
        y_lds[posl0 + r][o] = (acc[t][r] + bo) * mo;
    }
  }
  __syncthreads();

  // ---- decode: one thread per (pos, anchor) row, in place in LDS
  const int seg_lo[8] = {15, 19, 26, 47, 49, 51, 68, 71};
  const int seg_hi[8] = {18, 21, 28, 49, 51, 53, 71, 73};
  for (int row = tid; row < POSB * KNA; row += NTHREADS) {
    int posl = row & (POSB - 1);
    int a    = row >> 6;
    int p    = pbase + posl;
    if (p < HW) {
      float* yr = &y_lds[posl][a * KNO];
      float sv[18];
      int si = 0;
      #pragma unroll
      for (int s = 0; s < 8; ++s) {
        int lo = seg_lo[s], hi = seg_hi[s];
        float mx = yr[lo];
        for (int i = lo + 1; i < hi; ++i) mx = fmaxf(mx, yr[i]);
        float sum = 0.f;
        for (int i = lo; i < hi; ++i) {
          float e = __expf(yr[i] - mx);
          sv[si + (i - lo)] = e;
          sum += e;
        }
        float inv = 1.f / sum;
        for (int i = lo; i < hi; ++i) sv[si + (i - lo)] *= inv;
        si += hi - lo;
      }
      int gy = p / nxdim;
      int gx = p - gy * nxdim;
      float aw = (a == 0) ? aw0 : ((a == 1) ? aw1 : aw2);
      float ah = (a == 0) ? ah0 : ((a == 1) ? ah1 : ah2);
      float s0 = sigmoidf_(yr[0]);
      float s1 = sigmoidf_(yr[1]);
      float s2 = sigmoidf_(yr[2]);
      float s3 = sigmoidf_(yr[3]);
      float r0 = (s0 * 2.f - 0.5f + (float)gx) * stride;
      float r1 = (s1 * 2.f - 0.5f + (float)gy) * stride;
      float tw = s2 * 2.f, th = s3 * 2.f;
      float r2 = tw * tw * aw;
      float r3 = th * th * ah;
      for (int i = 4; i < KNO; ++i) yr[i] = sigmoidf_(yr[i]);
      si = 0;
      #pragma unroll
      for (int s = 0; s < 8; ++s)
        for (int i = seg_lo[s]; i < seg_hi[s]; ++i) yr[i] = sv[si++];
      yr[0] = r0; yr[1] = r1; yr[2] = r2; yr[3] = r3;
    }
  }
  __syncthreads();

  // ---- coalesced store: per anchor, rows (a*HW + pbase + posl) are contiguous
  size_t outbase = ((size_t)batch * (size_t)totalQ + (size_t)qoff) * KNO;
  int validPos = HW - pbase; if (validPos > POSB) validPos = POSB;
  int nElems = validPos * KNO;
  for (int a = 0; a < KNA; ++a) {
    size_t abase = outbase + ((size_t)a * HW + pbase) * (size_t)KNO;
    for (int i = tid; i < nElems; i += NTHREADS) {
      int posl = i / KNO;
      int no   = i - posl * KNO;
      out[abase + i] = y_lds[posl][a * KNO + no];
    }
  }
}

extern "C" void kernel_launch(void* const* d_in, const int* in_sizes, int n_in,
                              void* d_out, int out_size, void* d_ws, size_t ws_size,
                              hipStream_t stream) {
  (void)n_in; (void)out_size; (void)d_ws; (void)ws_size;
  // setup_inputs() dict order: x0,w0,b0,ia0,im0, x1,w1,b1,ia1,im1, x2,w2,b2,ia2,im2
  const float* x0  = (const float*)d_in[0];
  const float* w0  = (const float*)d_in[1];
  const float* b0  = (const float*)d_in[2];
  const float* ia0 = (const float*)d_in[3];
  const float* im0 = (const float*)d_in[4];
  const float* x1  = (const float*)d_in[5];
  const float* w1  = (const float*)d_in[6];
  const float* b1  = (const float*)d_in[7];
  const float* ia1 = (const float*)d_in[8];
  const float* im1 = (const float*)d_in[9];
  const float* x2  = (const float*)d_in[10];
  const float* w2  = (const float*)d_in[11];
  const float* b2  = (const float*)d_in[12];
  const float* ia2 = (const float*)d_in[13];
  const float* im2 = (const float*)d_in[14];
  float* out = (float*)d_out;

  const int HW0 = 80 * 80, HW1 = 40 * 40, HW2 = 20 * 20;
  int bs = in_sizes[0] / (256 * HW0);
  int totalQ = 3 * (HW0 + HW1 + HW2);   // 25200
  int qoff1 = 3 * HW0;                  // 19200
  int qoff2 = qoff1 + 3 * HW1;          // 24000

  dim3 g0((HW0 + POSB - 1) / POSB, bs);
  idetect_level_kernel<<<g0, NTHREADS, 0, stream>>>(
      x0, w0, b0, ia0, im0, out, 256, HW0, 80, 8.f,
      12.f, 16.f, 19.f, 36.f, 40.f, 28.f, 0, totalQ);

  dim3 g1((HW1 + POSB - 1) / POSB, bs);
  idetect_level_kernel<<<g1, NTHREADS, 0, stream>>>(
      x1, w1, b1, ia1, im1, out, 512, HW1, 40, 16.f,
      36.f, 75.f, 76.f, 55.f, 72.f, 146.f, qoff1, totalQ);

  dim3 g2((HW2 + POSB - 1) / POSB, bs);
  idetect_level_kernel<<<g2, NTHREADS, 0, stream>>>(
      x2, w2, b2, ia2, im2, out, 1024, HW2, 20, 32.f,
      142.f, 110.f, 192.f, 243.f, 459.f, 401.f, qoff2, totalQ);
}
